// ClassifyModel_11622181503729
// MI455X (gfx1250) — matrix operand order repaired and checked by reading, not compiled, not run on hardware
//
#include <hip/hip_runtime.h>
#include <hip/hip_bf16.h>

// Problem constants
#define BB 8
#define SS 512
#define DD 768
#define LL 200

typedef __attribute__((ext_vector_type(16))) _Float16 v16h;
typedef __attribute__((ext_vector_type(8)))  _Float16 v8h;
typedef __attribute__((ext_vector_type(4)))  _Float16 v4h;
typedef __attribute__((ext_vector_type(8)))  float    v8f;

// ---------------------------------------------------------------------------
// WMMA GEMM, 16x32 output tile per wave32 (one A fragment, two B fragments,
// two v_wmma_f32_16x16x32_f16 per K-chunk). f16 operands, f32 accumulate.
// LDS B-tile stored transposed [n][k] so each B fragment is one contiguous
// 32-byte (v16h) read; A fragment is two aligned v8h reads.
// Compile-time: BTRANS (B given as [N,K]), ADDC (C += ...), ACT (0 none,
// 1 leaky 0.2, 2 sigmoid), SCALE (A columns scaled by a_colscale[k]).
// ---------------------------------------------------------------------------
template<int BTRANS, int ADDC, int ACT, int SCALE>
__global__ __launch_bounds__(32) void gemm16(
    const float* __restrict__ A, int lda,
    const float* __restrict__ B, int ldb,
    float* __restrict__ C, int ldc,
    const float* __restrict__ bias,
    const float* __restrict__ a_colscale,
    int M, int N, int K)
{
    __shared__ __align__(32) _Float16 As[16][32];   // [m][k]
    __shared__ __align__(32) _Float16 Bs[32][32];   // [n][k]  (transposed!)

    const int m0   = blockIdx.x * 16;
    const int n0   = blockIdx.y * 32;
    const int lane = threadIdx.x;
    const int mloc = lane & 15;
    const int hi   = lane >> 4;

    v8f acc0 = {}, acc1 = {};

    for (int k0 = 0; k0 < K; k0 += 32) {
        const bool fullK = (k0 + 32 <= K);

        // ---- stage A tile 16x32 ----
        if (fullK && (m0 + 16 <= M)) {
            #pragma unroll
            for (int t = 0; t < 4; ++t) {
                int idx = t * 32 + lane;
                int m = idx >> 3, q = idx & 7;
                float4 f = *(const float4*)(A + (size_t)(m0 + m) * lda + k0 + q * 4);
                if (SCALE) {
                    float4 sc = *(const float4*)(a_colscale + k0 + q * 4);
                    f.x *= sc.x; f.y *= sc.y; f.z *= sc.z; f.w *= sc.w;
                }
                v4h h = { (_Float16)f.x, (_Float16)f.y, (_Float16)f.z, (_Float16)f.w };
                *(v4h*)&As[m][q * 4] = h;
            }
            if (k0 + 32 < K)
                __builtin_prefetch(A + (size_t)(m0 + mloc) * lda + k0 + 32, 0, 1);
        } else {
            for (int i = lane; i < 512; i += 32) {
                int m = i >> 5, kk = i & 31;
                int gm = m0 + m, gk = k0 + kk;
                float v = 0.0f;
                if (gm < M && gk < K) {
                    v = A[(size_t)gm * lda + gk];
                    if (SCALE) v *= a_colscale[gk];
                }
                As[m][kk] = (_Float16)v;
            }
        }

        // ---- stage B tile into Bs[n][k] ----
        if (fullK && (n0 + 32 <= N)) {
            if (BTRANS) {   // B is [N,K]: contiguous in k -> contiguous v4h stores
                #pragma unroll
                for (int t = 0; t < 8; ++t) {
                    int idx = t * 32 + lane;
                    int n = idx >> 3, q = idx & 7;
                    float4 f = *(const float4*)(B + (size_t)(n0 + n) * ldb + k0 + q * 4);
                    v4h h = { (_Float16)f.x, (_Float16)f.y, (_Float16)f.z, (_Float16)f.w };
                    *(v4h*)&Bs[n][q * 4] = h;
                }
            } else {        // B is [K,N]: contiguous in n -> 4 scalar stores
                #pragma unroll
                for (int t = 0; t < 8; ++t) {
                    int idx = t * 32 + lane;
                    int kk = idx >> 3, q = idx & 7;
                    float4 f = *(const float4*)(B + (size_t)(k0 + kk) * ldb + n0 + q * 4);
                    Bs[q * 4 + 0][kk] = (_Float16)f.x;
                    Bs[q * 4 + 1][kk] = (_Float16)f.y;
                    Bs[q * 4 + 2][kk] = (_Float16)f.z;
                    Bs[q * 4 + 3][kk] = (_Float16)f.w;
                }
                if (k0 + 32 < K)
                    __builtin_prefetch(B + (size_t)(k0 + 32 + lane) * ldb + n0, 0, 1);
            }
        } else {
            for (int i = lane; i < 1024; i += 32) {
                int kk = i >> 5, n = i & 31;
                int gk = k0 + kk, gn = n0 + n;
                float v = 0.0f;
                if (gk < K && gn < N)
                    v = BTRANS ? B[(size_t)gn * ldb + gk] : B[(size_t)gk * ldb + gn];
                Bs[n][kk] = (_Float16)v;
            }
        }
        __syncthreads();

        // ---- fragments: contiguous vector reads, same k = hi*16 + s order in A and B ----
        v8h a_lo = *(const v8h*)&As[mloc][hi * 16];
        v8h a_hi = *(const v8h*)&As[mloc][hi * 16 + 8];
        v16h af  = __builtin_shufflevector(a_lo, a_hi,
                       0,1,2,3,4,5,6,7,8,9,10,11,12,13,14,15);
        v16h bf0 = *(const v16h*)&Bs[mloc][hi * 16];
        v16h bf1 = *(const v16h*)&Bs[16 + mloc][hi * 16];

        acc0 = __builtin_amdgcn_wmma_f32_16x16x32_f16(
                   false, af, false, bf0, (short)0, acc0, false, false);
        acc1 = __builtin_amdgcn_wmma_f32_16x16x32_f16(
                   false, af, false, bf1, (short)0, acc1, false, false);
        __syncthreads();
    }

    const int gn0 = n0 + mloc;
    const int gn1 = n0 + 16 + mloc;
    float bv0 = 0.0f, bv1 = 0.0f;
    if (bias) {
        if (gn0 < N) bv0 = bias[gn0];
        if (gn1 < N) bv1 = bias[gn1];
    }

    #pragma unroll
    for (int r = 0; r < 8; ++r) {
        int gm = m0 + r + hi * 8;
        if (gm < M) {
            if (gn0 < N) {
                float v = acc0[r];
                if (ADDC) v += C[(size_t)gm * ldc + gn0];
                v += bv0;
                if (ACT == 1)      v = (v >= 0.0f) ? v : 0.2f * v;
                else if (ACT == 2) v = 1.0f / (1.0f + __expf(-v));
                C[(size_t)gm * ldc + gn0] = v;
            }
            if (gn1 < N) {
                float v = acc1[r];
                if (ADDC) v += C[(size_t)gm * ldc + gn1];
                v += bv1;
                if (ACT == 1)      v = (v >= 0.0f) ? v : 0.2f * v;
                else if (ACT == 2) v = 1.0f / (1.0f + __expf(-v));
                C[(size_t)gm * ldc + gn1] = v;
            }
        }
    }
}

// d[m] = sum_l A[l,m];  dinv[m] = d>0 ? rsqrt(d) : 0
__global__ void col_dinv_kernel(const float* __restrict__ AM, float* __restrict__ DV)
{
    int m = blockIdx.x * blockDim.x + threadIdx.x;
    if (m < LL) {
        float s = 0.0f;
        for (int l = 0; l < LL; ++l) s += AM[l * LL + m];
        DV[m] = (s > 0.0f) ? rsqrtf(s) : 0.0f;
    }
}

// A_norm[l,m] = A[m,l] * d[l] * d[m]
__global__ void anorm_kernel(const float* __restrict__ AM, const float* __restrict__ DV,
                             float* __restrict__ AN)
{
    int idx = blockIdx.x * blockDim.x + threadIdx.x;
    if (idx < LL * LL) {
        int l = idx / LL, m = idx % LL;
        AN[idx] = AM[m * LL + l] * DV[l] * DV[m];
    }
}

// L1-normalize columns (over L rows) of [L x D]
__global__ void l1norm_cols_kernel(const float* __restrict__ src, float* __restrict__ dst)
{
    int d = blockIdx.x * blockDim.x + threadIdx.x;
    if (d < DD) {
        float s = 0.0f;
        for (int l = 0; l < LL; ++l) s += fabsf(src[l * DD + d]);
        float r = 1.0f / fmaxf(s, 1e-12f);
        for (int l = 0; l < LL; ++l) dst[l * DD + d] = src[l * DD + d] * r;
    }
}

// inv[b,d] = 1 / max(sum_s |doc[b,s,d]|, 1e-12)
__global__ void doc_inv_kernel(const float* __restrict__ doc, float* __restrict__ INV)
{
    int idx = blockIdx.x * blockDim.x + threadIdx.x;
    if (idx < BB * DD) {
        int b = idx / DD, d = idx % DD;
        float s = 0.0f;
        for (int t = 0; t < SS; ++t) s += fabsf(doc[(b * SS + t) * DD + d]);
        INV[idx] = 1.0f / fmaxf(s, 1e-12f);
    }
}

// Conv weights -> WMMA-fragment-swizzled f16 matrix BM[13*3*32 lanes][16 halves],
// bias -> [208] with -3e38 padding.
// Fragment (ot, c, lane) element e:  k = c*32 + (lane>>4)*16 + e,
//                                    o = ot*16 + (lane&15)
__global__ void prep_convw_kernel(const float* __restrict__ conv_w,
                                  const float* __restrict__ conv_b,
                                  _Float16* __restrict__ Bmat, float* __restrict__ biasv)
{
    int i = blockIdx.x * blockDim.x + threadIdx.x;
    if (i < 13 * 3 * 32 * 16) {
        int e    = i & 15;
        int lane = (i >> 4) & 31;
        int rest = i >> 9;
        int c    = rest % 3;
        int ot   = rest / 3;
        int k    = c * 32 + (lane >> 4) * 16 + e;
        int o    = ot * 16 + (lane & 15);
        float v = 0.0f;
        if (k < 81 && o < LL) v = conv_w[o * 81 + k];
        Bmat[i] = (_Float16)v;
    }
    if (i < 208) biasv[i] = (i < LL) ? conv_b[i] : -3.0e38f;
}

// ---------------------------------------------------------------------------
// Fused conv(1->200ch, 9x9, pad 4) + bias + max over (channel, l) + tanh gate.
// Implicit GEMM: rows = l positions, cols = out channels, K = 81 taps (pad 96).
// One wave32 per (batch, s). Weight fragments pre-swizzled: one v16h per lane.
// ---------------------------------------------------------------------------
__global__ __launch_bounds__(32) void conv_max_gate_kernel(
    const float* __restrict__ wla, const v16h* __restrict__ Bmat,
    const float* __restrict__ biasv, float* __restrict__ gate)
{
    const int s    = blockIdx.x;
    const int b    = blockIdx.y;
    const int lane = threadIdx.x;
    const int mloc = lane & 15;
    const int hi   = lane >> 4;

    __shared__ _Float16 shw[9][224];      // wla rows s-4..s+4, halo-padded l
    __shared__ v16h     shB[13 * 3 * 32]; // fragment-order weights (39936 B)
    __shared__ float    red[32];

    for (int i = lane; i < 9 * 224; i += 32) {
        int dy = i / 224, c = i % 224;
        int sr = s + dy - 4;
        int ll = c - 4;
        float v = 0.0f;
        if (sr >= 0 && sr < SS && ll >= 0 && ll < LL)
            v = wla[(b * SS + sr) * LL + ll];
        shw[dy][c] = (_Float16)v;
    }
    {   // 16-byte vector copy: 13*3*32 v16h = 2496 uint4
        const uint4* src = (const uint4*)Bmat;
        uint4*       dst = (uint4*)&shB[0];
        for (int i = lane; i < 13 * 3 * 32 * 2; i += 32) dst[i] = src[i];
    }
    __syncthreads();

    float best = -3.0e38f;

    for (int lt = 0; lt < 13; ++lt) {
        // Build the 3 A fragments (taps 0..31, 32..63, 64..95) for this l-tile
        v16h af0 = {}, af1 = {}, af2 = {};
        const int l = lt * 16 + mloc;
        #pragma unroll
        for (int e = 0; e < 16; ++e) {
            int kb = hi * 16 + e;   // = af element -> tap index base
            {   // chunk 0: kk = kb (0..31)
                int kk = kb;
                af0[e] = shw[kk / 9][l + kk % 9];
            }
            {   // chunk 1
                int kk = 32 + kb;
                af1[e] = shw[kk / 9][l + kk % 9];
            }
            {   // chunk 2 (taps >= 81 are zero)
                int kk = 64 + kb;
                _Float16 v = (_Float16)0.0f;
                if (kk < 81) v = shw[kk / 9][l + kk % 9];
                af2[e] = v;
            }
        }

        for (int ot = 0; ot < 13; ++ot) {
            v8f acc = {};
            acc = __builtin_amdgcn_wmma_f32_16x16x32_f16(
                      false, af0, false, shB[(ot * 3 + 0) * 32 + lane], (short)0, acc, false, false);
            acc = __builtin_amdgcn_wmma_f32_16x16x32_f16(
                      false, af1, false, shB[(ot * 3 + 1) * 32 + lane], (short)0, acc, false, false);
            acc = __builtin_amdgcn_wmma_f32_16x16x32_f16(
                      false, af2, false, shB[(ot * 3 + 2) * 32 + lane], (short)0, acc, false, false);

            float bv = biasv[ot * 16 + mloc];   // padded channels: -3e38 -> excluded
            float tmax = acc[0] + bv;
            #pragma unroll
            for (int r = 1; r < 8; ++r) tmax = fmaxf(tmax, acc[r] + bv);
            // l-range mask is uniform per lane: only last tile's hi half is invalid
            if (lt < 12 || hi == 0) best = fmaxf(best, tmax);
        }
    }

    red[lane] = best;
    __syncthreads();
    if (lane == 0) {
        float m = red[0];
        for (int i = 1; i < 32; ++i) m = fmaxf(m, red[i]);
        gate[b * SS + s] = tanhf(m);
    }
}

// h_enc[b,d] = inv[b,d] * sum_s gate[b,s] * doc[b,s,d]
__global__ void henc_kernel(const float* __restrict__ doc, const float* __restrict__ INV,
                            const float* __restrict__ gate, float* __restrict__ H)
{
    int idx = blockIdx.x * blockDim.x + threadIdx.x;
    if (idx < BB * DD) {
        int b = idx / DD, d = idx % DD;
        float s = 0.0f;
        for (int t = 0; t < SS; ++t)
            s += gate[b * SS + t] * doc[(b * SS + t) * DD + d];
        H[idx] = s * INV[idx];
    }
}

// loss = -mean( y*log(p) + (1-y)*log(1-p) ), p clipped to [1e-7, 1-1e-7]
__global__ void loss_kernel(const float* __restrict__ p, const float* __restrict__ labels,
                            float* __restrict__ out)
{
    __shared__ float sh[256];
    int t = threadIdx.x;
    float s = 0.0f;
    const int n = BB * LL;
    for (int i = t; i < n; i += 256) {
        float pi = fminf(fmaxf(p[i], 1e-7f), 1.0f - 1e-7f);
        float y  = labels[i];
        s += y * logf(pi) + (1.0f - y) * logf(1.0f - pi);
    }
    sh[t] = s;
    __syncthreads();
    for (int off = 128; off > 0; off >>= 1) {
        if (t < off) sh[t] += sh[t + off];
        __syncthreads();
    }
    if (t == 0) *out = -sh[0] / (float)n;
}

// ---------------------------------------------------------------------------
extern "C" void kernel_launch(void* const* d_in, const int* in_sizes, int n_in,
                              void* d_out, int out_size, void* d_ws, size_t ws_size,
                              hipStream_t stream)
{
    (void)in_sizes; (void)n_in; (void)out_size; (void)ws_size;

    const float* doc       = (const float*)d_in[0];   // [B,S,D]
    const float* labels    = (const float*)d_in[1];   // [B,L]
    const float* label_emb = (const float*)d_in[2];   // [L,D]
    const float* label_adj = (const float*)d_in[3];   // [L,L]
    const float* w_gcn3    = (const float*)d_in[4];   // [D,D]
    const float* w_gcn5    = (const float*)d_in[5];   // [D,D]
    const float* w_ll1     = (const float*)d_in[6];   // [D,D]
    const float* b_ll1     = (const float*)d_in[7];   // [D]
    const float* w_ll2     = (const float*)d_in[8];   // [D,D]
    const float* b_ll2     = (const float*)d_in[9];   // [D]
    const float* w_lin     = (const float*)d_in[10];  // [2D,D]
    const float* b_lin     = (const float*)d_in[11];  // [D]
    const float* conv_w    = (const float*)d_in[12];  // [L,1,9,9]
    const float* conv_b    = (const float*)d_in[13];  // [L]
    const float* w_lin1    = (const float*)d_in[14];  // [D,L]
    const float* b_lin1    = (const float*)d_in[15];  // [L]

    float* out = (float*)d_out;  // [B*L] probs, then [1] loss

    // Workspace carve-up (floats; every block 16B-aligned)
    float* W     = (float*)d_ws;
    float* T     = W; W += LL * DD;           // scratch [L,D]
    float* LEb   = W; W += LL * DD;           // le
    float* Qb    = W; W += LL * DD;           // q / d_le
    float* Kb    = W; W += LL * DD;           // k / le_out
    float* AM    = W; W += LL * LL;           // sigmoid adjacency
    float* AN    = W; W += LL * LL;           // normalized adjacency
    float* DV    = W; W += 256;               // d^{-1/2}
    float* INV   = W; W += BB * DD;           // doc L1 inverse scales
    float* WLA   = W; W += BB * SS * LL;      // word-label attention
    float* GATE  = W; W += BB * SS;           // tanh gate
    float* H     = W; W += BB * DD;           // h_enc
    float* BIASV = W; W += 208;               // padded conv bias
    _Float16* BM = (_Float16*)W;              // swizzled f16 conv weights (19968 halves)

    const dim3 gLD((LL + 15) / 16, (DD + 31) / 32);  // 13 x 24
    const dim3 gLL((LL + 15) / 16, (LL + 31) / 32);  // 13 x 7

    // 1. t1 = label_emb @ w_gcn3                              [L,D]
    gemm16<0,0,0,0><<<gLD, 32, 0, stream>>>(label_emb, DD, w_gcn3, DD, T, DD,
                                            nullptr, nullptr, LL, DD, DD);
    // 2. le = leaky(label_adj @ t1)
    gemm16<0,0,1,0><<<gLD, 32, 0, stream>>>(label_adj, LL, T, DD, LEb, DD,
                                            nullptr, nullptr, LL, DD, LL);
    // 3. q = le @ w_ll1 + b_ll1
    gemm16<0,0,0,0><<<gLD, 32, 0, stream>>>(LEb, DD, w_ll1, DD, Qb, DD,
                                            b_ll1, nullptr, LL, DD, DD);
    // 4. k = le @ w_ll2 + b_ll2
    gemm16<0,0,0,0><<<gLD, 32, 0, stream>>>(LEb, DD, w_ll2, DD, Kb, DD,
                                            b_ll2, nullptr, LL, DD, DD);
    // 5. A = sigmoid(q @ k^T)                                 [L,L]
    gemm16<1,0,2,0><<<gLL, 32, 0, stream>>>(Qb, DD, Kb, DD, AM, LL,
                                            nullptr, nullptr, LL, LL, DD);
    // 6-7. degree normalize: A_norm = D A^T D
    col_dinv_kernel<<<1, 256, 0, stream>>>(AM, DV);
    anorm_kernel<<<(LL * LL + 255) / 256, 256, 0, stream>>>(AM, DV, AN);
    // 8. t2 = le @ w_gcn5
    gemm16<0,0,0,0><<<gLD, 32, 0, stream>>>(LEb, DD, w_gcn5, DD, T, DD,
                                            nullptr, nullptr, LL, DD, DD);
    // 9. d_le = leaky(A_norm @ t2)    (overwrites q)
    gemm16<0,0,1,0><<<gLD, 32, 0, stream>>>(AN, LL, T, DD, Qb, DD,
                                            nullptr, nullptr, LL, DD, LL);
    // 10. tmp = le @ w_lin[0:D,:]     (into Kb)
    gemm16<0,0,0,0><<<gLD, 32, 0, stream>>>(LEb, DD, w_lin, DD, Kb, DD,
                                            nullptr, nullptr, LL, DD, DD);
    // 11. le_out = d_le @ w_lin[D:2D,:] + tmp + b_lin
    gemm16<0,1,0,0><<<gLD, 32, 0, stream>>>(Qb, DD, w_lin + DD * DD, DD, Kb, DD,
                                            b_lin, nullptr, LL, DD, DD);
    // 12. le_n = l1norm_cols(le_out)  (into T)
    l1norm_cols_kernel<<<(DD + 255) / 256, 256, 0, stream>>>(Kb, T);
    // 13. doc L1 inverse scales
    doc_inv_kernel<<<(BB * DD + 255) / 256, 256, 0, stream>>>(doc, INV);
    // 14. wla[b] = (doc[b] * inv[b]) @ le_n^T                 [S,L] per batch
    for (int b = 0; b < BB; ++b) {
        gemm16<1,0,0,1><<<dim3(SS / 16, (LL + 31) / 32), 32, 0, stream>>>(
            doc + (size_t)b * SS * DD, DD, T, DD,
            WLA + (size_t)b * SS * LL, LL,
            nullptr, INV + b * DD, SS, LL, DD);
    }
    // 15. conv weight rearrange (fragment swizzle)
    prep_convw_kernel<<<(13 * 3 * 32 * 16 + 255) / 256, 256, 0, stream>>>(conv_w, conv_b, BM, BIASV);
    // 16. fused conv + bias + max(o,l) + tanh  -> gate[b,s]
    conv_max_gate_kernel<<<dim3(SS, BB), 32, 0, stream>>>(WLA, (const v16h*)BM, BIASV, GATE);
    // 17. h_enc
    henc_kernel<<<(BB * DD + 255) / 256, 256, 0, stream>>>(doc, INV, GATE, H);
    // 18. label_output = sigmoid(h_enc @ w_lin1 + b_lin1)  -> d_out[0:1600]
    gemm16<0,0,2,0><<<dim3(1, (LL + 31) / 32), 32, 0, stream>>>(H, DD, w_lin1, LL, out, LL,
                                                                b_lin1, nullptr, BB, LL, DD);
    // 19. BCE loss -> d_out[1600]
    loss_kernel<<<1, 256, 0, stream>>>(out, labels, out + BB * LL);
}
